// PretrainedModel_87797721465415
// MI455X (gfx1250) — matrix operand order repaired and checked by reading, not compiled, not run on hardware
//
#include <hip/hip_runtime.h>
#include <hip/hip_bf16.h>
#include <math.h>

typedef __attribute__((ext_vector_type(4)))  float   v4f;
typedef __attribute__((ext_vector_type(8)))  float   v8f;
typedef __attribute__((ext_vector_type(16))) __bf16  v16bf;

#define B_   32
#define S_   512
#define D_   768
#define N_   128
#define E_   128
#define L_   4
#define KN_  18     // n_node
#define KE_  14     // n_edge

// ---------------------------------------------------------------------------
// Node path: fused gather + masked mean + GEMM(bf16 WMMA, f32 acc) + bias+tanh
// One wave per 16-row M-tile. M = B*N = 4096 -> 256 tiles -> 64 blocks x 4 waves.
// Masked spans are skipped via EXEC-masked loads (no wasted HBM traffic).
// ---------------------------------------------------------------------------
__global__ __launch_bounds__(128) void node_fused_kernel(
    const float* __restrict__ hidden,     // [B,S,D]
    const int*   __restrict__ node_idx,   // [B,N,L]
    const int*   __restrict__ node_mask,  // [B,N,L]
    const float* __restrict__ W_node,     // [KN_, D]
    const float* __restrict__ b_node,     // [KN_]
    float*       __restrict__ out)        // [B,N,KN_]
{
    const int lane = threadIdx.x & 31;
    const int tile = blockIdx.x * (blockDim.x >> 5) + (threadIdx.x >> 5);
    const int m0   = tile * 16;
    const int r    = m0 + (lane & 15);          // span row in [0, B*N)
    const int bi   = r / N_;

    // Per-lane span gather state (lanes 16..31 mirror 0..15's rows).
    const float* ptr[L_];
    bool  act[L_];
    float cnt = 0.f;
#pragma unroll
    for (int l = 0; l < L_; ++l) {
        const int idx = node_idx [r * L_ + l];
        const int m   = node_mask[r * L_ + l];
        ptr[l] = hidden + ((size_t)bi * S_ + (size_t)idx) * D_;
        act[l] = (m != 0);
        cnt   += (float)m;
    }
    const float inv   = 1.0f / fmaxf(cnt, 1.0f);
    const int   off   = (lane < 16) ? 0 : 8;    // A-fragment K sub-offset (ISA layout)
    const int   koffB = (lane < 16) ? 0 : 8;    // B-fragment K sub-offset (ISA layout)
    const int   ncol  = lane & 15;

    v8f acc0 = {0.f,0.f,0.f,0.f,0.f,0.f,0.f,0.f};   // cols 0..15
    v8f acc1 = {0.f,0.f,0.f,0.f,0.f,0.f,0.f,0.f};   // cols 16..17 (rest padded)

    for (int kb = 0; kb < D_; kb += 32) {
        const int kA = kb + off;
        // ---- A fragment: EXEC-masked span accumulate, mean, cvt to bf16 ----
        v4f s0 = {0.f,0.f,0.f,0.f};   // K = kA .. kA+3
        v4f s1 = {0.f,0.f,0.f,0.f};   // K = kA+4 .. kA+7
        v4f s2 = {0.f,0.f,0.f,0.f};   // K = kA+16 .. kA+19
        v4f s3 = {0.f,0.f,0.f,0.f};   // K = kA+20 .. kA+23
#pragma unroll
        for (int l = 0; l < L_; ++l) {
            if (act[l]) {                               // masked lanes skip loads
                const v4f* p = (const v4f*)(ptr[l] + kA);   // 32B aligned
                s0 += p[0];  s1 += p[1];
                s2 += p[4];  s3 += p[5];
            }
        }
        s0 *= inv;  s1 *= inv;  s2 *= inv;  s3 *= inv;
        v16bf a;
#pragma unroll
        for (int j = 0; j < 4; ++j) {
            a[j]      = (__bf16)s0[j];
            a[4 + j]  = (__bf16)s1[j];
            a[8 + j]  = (__bf16)s2[j];
            a[12 + j] = (__bf16)s3[j];
        }
        // ---- B fragments: W_node^T tiles, zero-pad cols >= 18 ----
        v16bf bf0, bf1;
        {
            const v4f* wp = (const v4f*)(W_node + (size_t)ncol * D_ + kb + koffB);
#pragma unroll
            for (int q = 0; q < 4; ++q) {
                const v4f wv = wp[q + (q & 2)];
#pragma unroll
                for (int j = 0; j < 4; ++j) bf0[4 * q + j] = (__bf16)wv[j];
            }
        }
        {
            const int n2 = ncol + 16;
            if (n2 < KN_) {
                const v4f* wp = (const v4f*)(W_node + (size_t)n2 * D_ + kb + koffB);
#pragma unroll
                for (int q = 0; q < 4; ++q) {
                    const v4f wv = wp[q + (q & 2)];
#pragma unroll
                    for (int j = 0; j < 4; ++j) bf1[4 * q + j] = (__bf16)wv[j];
                }
            } else {
#pragma unroll
                for (int e = 0; e < 16; ++e) bf1[e] = (__bf16)0.f;
            }
        }
        acc0 = __builtin_amdgcn_wmma_f32_16x16x32_bf16(false, a, false, bf0,
                                                       (short)0, acc0, false, false);
        acc1 = __builtin_amdgcn_wmma_f32_16x16x32_bf16(false, a, false, bf1,
                                                       (short)0, acc1, false, false);
    }
    // ---- epilogue: bias + tanh*5, masked column writes ----
#pragma unroll
    for (int v = 0; v < 8; ++v) {
        const int m = m0 + v + 8 * (lane >> 4);
        out[m * KN_ + ncol] = tanhf(acc0[v] + b_node[ncol]) * 5.0f;
        const int n2 = ncol + 16;
        if (n2 < KN_)
            out[m * KN_ + n2] = tanhf(acc1[v] + b_node[n2]) * 5.0f;
    }
}

// ---------------------------------------------------------------------------
// Edge path: row = concat(head-span mean, tail-span mean) -> K = 1536.
// K loop split per span so all register-array indexing is static.
// M = B*E = 4096 -> 256 tiles -> 64 blocks x 4 waves. Single N-tile (14 cols).
// ---------------------------------------------------------------------------
__global__ __launch_bounds__(128) void edge_fused_kernel(
    const float* __restrict__ hidden,     // [B,S,D]
    const int*   __restrict__ edge_idx,   // [B,E,2,L]
    const int*   __restrict__ edge_mask,  // [B,E,2,L]
    const float* __restrict__ W_edge,     // [KE_, 2*D]
    const float* __restrict__ b_edge,     // [KE_]
    float*       __restrict__ out)        // [B,E,KE_]
{
    const int lane = threadIdx.x & 31;
    const int tile = blockIdx.x * (blockDim.x >> 5) + (threadIdx.x >> 5);
    const int m0   = tile * 16;
    const int r    = m0 + (lane & 15);          // edge row in [0, B*E)
    const int bi   = r / E_;

    const float* ptr[2][L_];
    bool  act[2][L_];
    float inv[2];
#pragma unroll
    for (int s = 0; s < 2; ++s) {
        float cnt = 0.f;
#pragma unroll
        for (int l = 0; l < L_; ++l) {
            const int base = ((r * 2) + s) * L_ + l;
            const int idx  = edge_idx [base];
            const int m    = edge_mask[base];
            ptr[s][l] = hidden + ((size_t)bi * S_ + (size_t)idx) * D_;
            act[s][l] = (m != 0);
            cnt      += (float)m;
        }
        inv[s] = 1.0f / fmaxf(cnt, 1.0f);
    }
    const int off   = (lane < 16) ? 0 : 8;
    const int koffB = (lane < 16) ? 0 : 8;
    const int ncol  = lane & 15;
    const bool colOK = (ncol < KE_);

    v8f acc = {0.f,0.f,0.f,0.f,0.f,0.f,0.f,0.f};

#pragma unroll
    for (int s = 0; s < 2; ++s) {               // fully unrolled: static indexing
        const float ivs = inv[s];
        for (int kb = 0; kb < D_; kb += 32) {
            const int kA = kb + off;
            // ---- A fragment ----
            v4f s0 = {0.f,0.f,0.f,0.f};
            v4f s1 = {0.f,0.f,0.f,0.f};
            v4f s2 = {0.f,0.f,0.f,0.f};
            v4f s3 = {0.f,0.f,0.f,0.f};
#pragma unroll
            for (int l = 0; l < L_; ++l) {
                if (act[s][l]) {
                    const v4f* p = (const v4f*)(ptr[s][l] + kA);
                    s0 += p[0];  s1 += p[1];
                    s2 += p[4];  s3 += p[5];
                }
            }
            s0 *= ivs;  s1 *= ivs;  s2 *= ivs;  s3 *= ivs;
            v16bf a;
#pragma unroll
            for (int j = 0; j < 4; ++j) {
                a[j]      = (__bf16)s0[j];
                a[4 + j]  = (__bf16)s1[j];
                a[8 + j]  = (__bf16)s2[j];
                a[12 + j] = (__bf16)s3[j];
            }
            // ---- B fragment: W_edge^T (K offset s*D_), zero-pad cols >= 14 ----
            v16bf bf;
            if (colOK) {
                const v4f* wp =
                    (const v4f*)(W_edge + (size_t)ncol * (2 * D_) + s * D_ + kb + koffB);
#pragma unroll
                for (int q = 0; q < 4; ++q) {
                    const v4f wv = wp[q + (q & 2)];
#pragma unroll
                    for (int j = 0; j < 4; ++j) bf[4 * q + j] = (__bf16)wv[j];
                }
            } else {
#pragma unroll
                for (int e = 0; e < 16; ++e) bf[e] = (__bf16)0.f;
            }
            acc = __builtin_amdgcn_wmma_f32_16x16x32_bf16(false, a, false, bf,
                                                          (short)0, acc, false, false);
        }
    }
#pragma unroll
    for (int v = 0; v < 8; ++v) {
        const int m = m0 + v + 8 * (lane >> 4);
        if (colOK)
            out[m * KE_ + ncol] = tanhf(acc[v] + b_edge[ncol]) * 5.0f;
    }
}

extern "C" void kernel_launch(void* const* d_in, const int* in_sizes, int n_in,
                              void* d_out, int out_size, void* d_ws, size_t ws_size,
                              hipStream_t stream) {
    (void)in_sizes; (void)n_in; (void)out_size; (void)d_ws; (void)ws_size;
    const float* hidden    = (const float*)d_in[0];
    const int*   node_idx  = (const int*)  d_in[1];
    const int*   node_mask = (const int*)  d_in[2];
    const int*   edge_idx  = (const int*)  d_in[3];
    const int*   edge_mask = (const int*)  d_in[4];
    const float* W_node    = (const float*)d_in[5];
    const float* b_node    = (const float*)d_in[6];
    const float* W_edge    = (const float*)d_in[7];
    const float* b_edge    = (const float*)d_in[8];

    float* node_out = (float*)d_out;                 // [32,128,18]
    float* edge_out = node_out + B_ * N_ * KN_;      // [32,128,14]

    // 256 M-tiles each; 4 waves (128 threads) per block -> 64 blocks per kernel.
    node_fused_kernel<<<64, 128, 0, stream>>>(hidden, node_idx, node_mask,
                                              W_node, b_node, node_out);
    edge_fused_kernel<<<64, 128, 0, stream>>>(hidden, edge_idx, edge_mask,
                                              W_edge, b_edge, edge_out);
}
